// MultiHeadCausalAttention_82961588290058
// MI455X (gfx1250) — matrix operand order repaired and checked by reading, not compiled, not run on hardware
//
#include <hip/hip_runtime.h>

// ---------------- types ----------------
typedef __bf16 v16bf __attribute__((ext_vector_type(16)));
typedef float  v8f   __attribute__((ext_vector_type(8)));
typedef unsigned int v4u __attribute__((ext_vector_type(4)));

union FragU { v16bf v; v4u q[2]; };

__device__ __forceinline__ unsigned short f2bf(float f) {
    unsigned int u = __float_as_uint(f);
    u += 0x7fffu + ((u >> 16) & 1u);           // round-to-nearest-even
    return (unsigned short)(u >> 16);
}

__device__ __forceinline__ v8f wmma_bf16(v16bf a, v16bf b, v8f c) {
    return __builtin_amdgcn_wmma_f32_16x16x32_bf16(false, a, false, b, (short)0, c, false, false);
}

// A fragment (16x32), matrix stored row-major (stride in elems).
// lane<16: row=r, K = {k0..k0+7, k0+16..k0+23}; lane>=16: +8 on k base.
__device__ __forceinline__ v16bf frag_a_any(const unsigned short* m, int stride,
                                            int row0, int k0, int lane) {
    int r  = row0 + (lane & 15);
    int kb = k0 + ((lane & 16) ? 8 : 0);
    const v4u* p = (const v4u*)(m + (size_t)r * stride + kb);
    FragU f; f.q[0] = p[0]; f.q[1] = p[2];      // +16 elems = +2 v4u
    return f.v;
}

// B fragment (32x16), matrix stored TRANSPOSED (rows = logical columns).
// lane<16: col=n, K = {k0..k0+7, k0+16..k0+23}; lane>=16: +8 on k base.
__device__ __forceinline__ v16bf frag_b_any(const unsigned short* mt, int stride,
                                            int n0, int k0, int lane) {
    int n  = n0 + (lane & 15);
    int kb = k0 + ((lane & 16) ? 8 : 0);
    const v4u* p = (const v4u*)(mt + (size_t)n * stride + kb);
    FragU f; f.q[0] = p[0]; f.q[1] = p[2];
    return f.v;
}

// ---------------- constants ----------------
#define NROWS 2048
#define DDIM  1024
#define QVKC  3072
#define HEADS 16
#define EDIM  64
#define CHUNK 64
#define NCH   32
#define SCOLS 80            // 64 value cols + 1 ones col (z) + 15 pad
#define SSZ   (SCOLS * EDIM)  // 5120 f32 per (head,chunk) state

// ---------------- conversion kernels ----------------
__global__ void cvt_x_kernel(const float* __restrict__ x, unsigned short* __restrict__ xb) {
    size_t i = (size_t)blockIdx.x * blockDim.x + threadIdx.x;
    xb[i] = f2bf(x[i]);
}

// W (1024 x 3072) row-major  ->  Wt (3072 x 1024) bf16 row-major
__global__ void cvt_wt_kernel(const float* __restrict__ W, unsigned short* __restrict__ wt) {
    size_t i = (size_t)blockIdx.x * blockDim.x + threadIdx.x;   // over 3072*1024
    int n = (int)(i >> 10);          // output column (0..3071)
    int k = (int)(i & 1023);         // reduction index
    wt[i] = f2bf(W[(size_t)k * QVKC + n]);
}

// ---------------- QKV projection GEMM (bf16 WMMA, f32 accum) ----------------
// C(2048x3072) = X(2048x1024) @ W(1024x3072) + b ; per-wave 64x64 tile.
__global__ __launch_bounds__(256) void gemm_qvk_kernel(
    const unsigned short* __restrict__ xb, const unsigned short* __restrict__ wt,
    const float* __restrict__ bias, float* __restrict__ qvk) {
    int tid = threadIdx.x, lane = tid & 31, w = tid >> 5;
    int wr = w >> 2, wc = w & 3;
    int row0 = blockIdx.y * 128 + wr * 64;
    int col0 = blockIdx.x * 256 + wc * 64;

    v8f acc[4][4];
#pragma unroll
    for (int i = 0; i < 4; i++)
#pragma unroll
        for (int j = 0; j < 4; j++) acc[i][j] = (v8f)0.f;

    for (int k0 = 0; k0 < DDIM; k0 += 32) {
        v16bf a[4], b[4];
#pragma unroll
        for (int i = 0; i < 4; i++) a[i] = frag_a_any(xb, DDIM, row0 + 16 * i, k0, lane);
#pragma unroll
        for (int j = 0; j < 4; j++) b[j] = frag_b_any(wt, DDIM, col0 + 16 * j, k0, lane);
#pragma unroll
        for (int i = 0; i < 4; i++)
#pragma unroll
            for (int j = 0; j < 4; j++) acc[i][j] = wmma_bf16(a[i], b[j], acc[i][j]);
    }

    int cl = lane & 15, rh = (lane & 16) ? 8 : 0;
    float bv[4];
#pragma unroll
    for (int j = 0; j < 4; j++) bv[j] = bias[col0 + 16 * j + cl];
#pragma unroll
    for (int i = 0; i < 4; i++)
#pragma unroll
        for (int j = 0; j < 4; j++)
#pragma unroll
            for (int v = 0; v < 8; v++)
                qvk[(size_t)(row0 + 16 * i + v + rh) * QVKC + col0 + 16 * j + cl] =
                    acc[i][j][v] + bv[j];
}

// ---------------- Pass A: per-chunk state G_t = exp(K)^T @ [V | 1 | 0] ----------------
__global__ __launch_bounds__(128) void passA_kernel(const float* __restrict__ qvk,
                                                    float* __restrict__ G) {
    int t = blockIdx.x, h = blockIdx.y;
    __shared__ __align__(16) unsigned short Kt[EDIM * CHUNK];   // [d][i]
    __shared__ __align__(16) unsigned short Vt[SCOLS * CHUNK];  // [c][i]
    int tid = threadIdx.x;

    if (tid < 64) {
        int i = tid;
        const float* krow = qvk + (size_t)(t * CHUNK + i) * QVKC + 2048 + h * EDIM;
#pragma unroll
        for (int d = 0; d < EDIM; d++) Kt[d * CHUNK + i] = f2bf(__expf(krow[d]));
    } else {
        int i = tid - 64;
        const float* vrow = qvk + (size_t)(t * CHUNK + i) * QVKC + 1024 + h * EDIM;
#pragma unroll
        for (int c = 0; c < EDIM; c++) Vt[c * CHUNK + i] = f2bf(vrow[c]);
        Vt[64 * CHUNK + i] = 0x3F80;           // ones column -> z state
#pragma unroll
        for (int c = 65; c < SCOLS; c++) Vt[c * CHUNK + i] = 0;
    }
    __syncthreads();

    int lane = tid & 31, w = tid >> 5, r0 = 16 * w;
    v8f acc[5];
#pragma unroll
    for (int j = 0; j < 5; j++) acc[j] = (v8f)0.f;
#pragma unroll
    for (int kk = 0; kk < 2; kk++) {
        v16bf a = frag_a_any(Kt, CHUNK, r0, kk * 32, lane);
#pragma unroll
        for (int j = 0; j < 5; j++)
            acc[j] = wmma_bf16(a, frag_b_any(Vt, CHUNK, 16 * j, kk * 32, lane), acc[j]);
    }
    // store transposed [c][d] so pass C can B-fragment it directly
    float* g = G + (size_t)(h * NCH + t) * SSZ;
    int cl = lane & 15, rh = (lane & 16) ? 8 : 0;
#pragma unroll
    for (int j = 0; j < 5; j++)
#pragma unroll
        for (int v = 0; v < 8; v++)
            g[(16 * j + cl) * EDIM + r0 + v + rh] = acc[j][v];
}

// ---------------- Pass B: exclusive prefix over chunks (per head) ----------------
__global__ __launch_bounds__(256) void passB_kernel(float* __restrict__ G) {
    int h = blockIdx.x, tid = threadIdx.x;
    const int PER = SSZ / 256;                 // 20
    float run[PER];
#pragma unroll
    for (int i = 0; i < PER; i++) run[i] = 0.f;
    size_t hb = (size_t)h * NCH * SSZ;
    for (int t = 0; t < NCH; t++) {
#pragma unroll
        for (int i = 0; i < PER; i++) {
            size_t idx = hb + (size_t)t * SSZ + tid + i * 256;
            float g = G[idx];
            G[idx]  = run[i];                  // exclusive prefix
            run[i] += g;
        }
    }
}

// ---------------- Pass C: O = mask(QK^T) @ V_aug + Q @ S_prefix_aug ----------------
__global__ __launch_bounds__(128) void passC_kernel(const float* __restrict__ qvk,
                                                    const float* __restrict__ G,
                                                    float* __restrict__ out) {
    int t = blockIdx.x, h = blockIdx.y;
    __shared__ __align__(16) unsigned short Qs[CHUNK * EDIM];   // row-major
    __shared__ __align__(16) unsigned short Ks[CHUNK * EDIM];   // row-major (= K^T col-major)
    __shared__ __align__(16) unsigned short Vt[SCOLS * CHUNK];  // [c][i]
    __shared__ __align__(16) unsigned short Ps[CHUNK * CHUNK];  // masked QK^T, bf16
    __shared__ __align__(16) unsigned short St[SCOLS * EDIM];   // S_prefix transposed, bf16
    __shared__ float denomS[CHUNK];
    int tid = threadIdx.x;

    // stage prefix state
    {
        const float* g = G + (size_t)(h * NCH + t) * SSZ;
        for (int e = tid; e < SSZ; e += 128) St[e] = f2bf(g[e]);
    }
    if (tid < 64) {
        int i = tid;
        const float* qrow = qvk + (size_t)(t * CHUNK + i) * QVKC + h * EDIM;
        float qv[EDIM];
        float mx = -3.4e38f;
#pragma unroll
        for (int d = 0; d < EDIM; d++) { qv[d] = qrow[d]; mx = fmaxf(mx, qv[d]); }
        float s = 0.f;
#pragma unroll
        for (int d = 0; d < EDIM; d++) { qv[d] = __expf(qv[d] - mx); s += qv[d]; }
        float sc = 0.125f / s;                 // softmax * e^-0.5
#pragma unroll
        for (int d = 0; d < EDIM; d++) Qs[i * EDIM + d] = f2bf(qv[d] * sc);
        const float* vrow = qvk + (size_t)(t * CHUNK + i) * QVKC + 1024 + h * EDIM;
#pragma unroll
        for (int c = 0; c < EDIM; c++) Vt[c * CHUNK + i] = f2bf(vrow[c]);
    } else {
        int i = tid - 64;
        const float* krow = qvk + (size_t)(t * CHUNK + i) * QVKC + 2048 + h * EDIM;
#pragma unroll
        for (int d = 0; d < EDIM; d++) Ks[i * EDIM + d] = f2bf(__expf(krow[d]));
        Vt[64 * CHUNK + i] = 0x3F80;
#pragma unroll
        for (int c = 65; c < SCOLS; c++) Vt[c * CHUNK + i] = 0;
    }
    __syncthreads();

    int lane = tid & 31, w = tid >> 5, r0 = 16 * w;
    int cl = lane & 15, rh = (lane & 16) ? 8 : 0;

    // P = Q @ K^T  (K row-major == K^T column-major -> B fragments)
    v16bf qf[2];
    qf[0] = frag_a_any(Qs, EDIM, r0, 0, lane);
    qf[1] = frag_a_any(Qs, EDIM, r0, 32, lane);
    v8f p[4];
#pragma unroll
    for (int j = 0; j < 4; j++) p[j] = (v8f)0.f;
#pragma unroll
    for (int kk = 0; kk < 2; kk++)
#pragma unroll
        for (int j = 0; j < 4; j++)
            p[j] = wmma_bf16(qf[kk], frag_b_any(Ks, EDIM, 16 * j, kk * 32, lane), p[j]);

    // causal mask (inclusive diagonal), spill masked P to LDS as bf16
#pragma unroll
    for (int j = 0; j < 4; j++)
#pragma unroll
        for (int v = 0; v < 8; v++) {
            int m = r0 + v + rh, n = 16 * j + cl;
            Ps[m * CHUNK + n] = f2bf((n <= m) ? p[j][v] : 0.f);
        }
    __syncthreads();

    // O = P @ V_aug + Q @ S_aug  (col 64 of O = denominator)
    v16bf pf[2];
    pf[0] = frag_a_any(Ps, CHUNK, r0, 0, lane);
    pf[1] = frag_a_any(Ps, CHUNK, r0, 32, lane);
    v8f o[5];
#pragma unroll
    for (int j = 0; j < 5; j++) o[j] = (v8f)0.f;
#pragma unroll
    for (int kk = 0; kk < 2; kk++)
#pragma unroll
        for (int j = 0; j < 5; j++)
            o[j] = wmma_bf16(pf[kk], frag_b_any(Vt, CHUNK, 16 * j, kk * 32, lane), o[j]);
#pragma unroll
    for (int kk = 0; kk < 2; kk++)
#pragma unroll
        for (int j = 0; j < 5; j++)
            o[j] = wmma_bf16(qf[kk], frag_b_any(St, EDIM, 16 * j, kk * 32, lane), o[j]);

    if (cl == 0) {
#pragma unroll
        for (int v = 0; v < 8; v++) denomS[r0 + v + rh] = o[4][v];
    }
    __syncthreads();

    float dn[8];
#pragma unroll
    for (int v = 0; v < 8; v++) dn[v] = denomS[r0 + v + rh] + 1.25e-7f;  // + eps*sum(q)

#pragma unroll
    for (int j = 0; j < 4; j++)
#pragma unroll
        for (int v = 0; v < 8; v++)
            out[(size_t)(t * CHUNK + r0 + v + rh) * DDIM + h * EDIM + 16 * j + cl] =
                o[j][v] / dn[v];
}

// ---------------- host ----------------
extern "C" void kernel_launch(void* const* d_in, const int* in_sizes, int n_in,
                              void* d_out, int out_size, void* d_ws, size_t ws_size,
                              hipStream_t stream) {
    (void)in_sizes; (void)n_in; (void)out_size; (void)ws_size;
    const float* x    = (const float*)d_in[0];
    const float* W    = (const float*)d_in[1];
    const float* bias = (const float*)d_in[2];
    float* out = (float*)d_out;

    unsigned short* xb  = (unsigned short*)d_ws;                      // 2048*1024 bf16
    unsigned short* wt  = xb + (size_t)NROWS * DDIM;                  // 3072*1024 bf16
    float* qvk = (float*)(wt + (size_t)QVKC * DDIM);                  // 2048*3072 f32
    float* G   = qvk + (size_t)NROWS * QVKC;                          // 16*32*5120 f32

    cvt_x_kernel<<<(NROWS * DDIM) / 256, 256, 0, stream>>>(x, xb);
    cvt_wt_kernel<<<(QVKC * DDIM) / 256, 256, 0, stream>>>(W, wt);
    gemm_qvk_kernel<<<dim3(QVKC / 256, NROWS / 128), 256, 0, stream>>>(xb, wt, bias, qvk);
    passA_kernel<<<dim3(NCH, HEADS), 128, 0, stream>>>(qvk, G);
    passB_kernel<<<HEADS, 256, 0, stream>>>(G);
    passC_kernel<<<dim3(NCH, HEADS), 128, 0, stream>>>(qvk, G, out);
}
